// EdgeConv_16655883174087
// MI455X (gfx1250) — compile-verified, matrix operand order checked by reading, not run on hardware
//
#include <hip/hip_runtime.h>

// ---------------- problem constants ----------------
#define B_    2
#define CH    64
#define NPTS  8192
#define KNN   32
#define COLS  (B_ * NPTS * KNN)   // 524288 edge columns
#define F0IN  128                 // layer0 input channels (2C)
#define F1    64                  // layer0 out
#define F2    64                  // layer1 out
#define F3    128                 // layer2 out
#define RT    64                  // gram row tile
#define CT    64                  // gram col tile
#define CT0   64                  // gemm column tile
#define NEG_SLOPE 0.2f

typedef __attribute__((ext_vector_type(16))) _Float16 v16h;
typedef __attribute__((ext_vector_type(8)))  float    v8f;

// -------- CDNA5 async global->LDS copy (ASYNCcnt path), with fallback ------
#if defined(__has_builtin)
#  if __has_builtin(__builtin_amdgcn_global_load_async_to_lds_b128)
#    define HAVE_ASYNC_LDS 1
#  endif
#  if __has_builtin(__builtin_amdgcn_s_wait_asynccnt)
#    define HAVE_WAIT_ASYNC 1
#  endif
#endif

// exact pointee type per hipcc diagnostic: vector_size(16) int, AS1 / AS3
typedef int v4i_gcc __attribute__((vector_size(16)));
typedef __attribute__((address_space(1))) v4i_gcc gv4i;
typedef __attribute__((address_space(3))) v4i_gcc lv4i;

__device__ __forceinline__ void cp16_async(void* dst, const void* src) {
#ifdef HAVE_ASYNC_LDS
  __builtin_amdgcn_global_load_async_to_lds_b128((gv4i*)src, (lv4i*)dst, 0, 0);
#else
  *(int4*)dst = *(const int4*)src;
#endif
}

__device__ __forceinline__ void wait_async0() {
#ifdef HAVE_ASYNC_LDS
#  ifdef HAVE_WAIT_ASYNC
  __builtin_amdgcn_s_wait_asynccnt(0);
#  else
  asm volatile("s_wait_asynccnt 0" ::: "memory");
#  endif
#endif
}

// issue async copies of `bytes` (16B-aligned) with 256 threads
__device__ __forceinline__ void stage_async_lds(void* dst, const void* src,
                                                int bytes, int tid) {
  for (int off = tid * 16; off < bytes; off += 256 * 16)
    cp16_async((char*)dst + off, (const char*)src + off);
}

// Per-lane gather of a 16x32 (f16) WMMA operand fragment from an LDS tile
// stored row-major with stride `rowStride` halves.
// CDNA5 16-bit A/B layout: lanes 0-15 hold K {0..7,16..23}, lanes 16-31 hold
// K {8..15,24..31} of the 32-wide K chunk; (lane&15) selects the row (M or N).
__device__ __forceinline__ v16h frag_ld(const _Float16* base, int rowStride,
                                        int row16, int kbase, int lane) {
  union { v16h v; _Float16 h[16]; } u;
  const _Float16* p = base + (size_t)(row16 + (lane & 15)) * rowStride
                    + kbase + ((lane & 16) ? 8 : 0);
#pragma unroll
  for (int i = 0; i < 8; ++i) { u.h[i] = p[i]; u.h[8 + i] = p[16 + i]; }
  return u.v;
}

__device__ __forceinline__ v8f wmma_f16(v16h a, v16h b, v8f c) {
  return __builtin_amdgcn_wmma_f32_16x16x32_f16(false, a, false, b,
                                                (short)0, c, false, false);
}

// ---------------- kernel 0: zero stats ----------------
__global__ void k_zero(float* __restrict__ p) {
  int t = threadIdx.x;
  p[t] = 0.f; p[t + 256] = 0.f; p[t + 512] = 0.f; p[t + 768] = 0.f;
}

// ---------------- kernel 0b: fp32 weights -> fp16 (once) -------------------
__global__ __launch_bounds__(256) void k_prep_w(const float* __restrict__ W0,
                                                const float* __restrict__ W1,
                                                const float* __restrict__ W2,
                                                _Float16* __restrict__ h0,
                                                _Float16* __restrict__ h1,
                                                _Float16* __restrict__ h2) {
  int t = blockIdx.x * 256 + threadIdx.x;
  if (t < F1 * F0IN) h0[t] = (_Float16)W0[t];
  if (t < F2 * F1)   h1[t] = (_Float16)W1[t];
  if (t < F3 * F2)   h2[t] = (_Float16)W2[t];
}

// ---------------- kernel 1: fp32 (B,C,N) -> fp16 point-major (B,N,C) + |p|^2
__global__ __launch_bounds__(256) void k_prep(const float* __restrict__ x,
                                              _Float16* __restrict__ P16,
                                              float* __restrict__ sq) {
  int t = blockIdx.x * 256 + threadIdx.x;  // global point id over B*N
  if (t >= B_ * NPTS) return;
  int b = t >> 13, n = t & (NPTS - 1);
  const float* xb = x + (size_t)b * CH * NPTS + n;
  float s = 0.f;
  _Float16 row[CH];
#pragma unroll 8
  for (int c = 0; c < CH; ++c) {
    float v = xb[(size_t)c * NPTS];
    s += v * v;
    row[c] = (_Float16)v;
  }
  sq[t] = s;
  _Float16* dst = P16 + (size_t)t * CH;
#pragma unroll 8
  for (int c = 0; c < CH; ++c) dst[c] = row[c];
}

// ------ kernel 2: fused Gram GEMM (WMMA) + running top-32, async double-buf -
__global__ __launch_bounds__(256) void k_gram_topk(const _Float16* __restrict__ P16,
                                                   const float* __restrict__ sq,
                                                   int* __restrict__ idxOut) {
  __shared__ __align__(16) _Float16 As[RT][CH];        // 8 KB
  __shared__ __align__(16) _Float16 Bs[2][CT][CH];     // 16 KB (double buffer)
  __shared__ float Ds[RT][CT + 1];                     // ~16 KB
  __shared__ float topd[RT][KNN];                      // 8 KB
  __shared__ int   topi[RT][KNN];                      // 8 KB
  __shared__ float sqr[RT];

  int b = blockIdx.y;
  int row0 = blockIdx.x * RT;
  int tid = threadIdx.x, lane = tid & 31, wave = tid >> 5;
  const _Float16* Pb = P16 + (size_t)b * NPTS * CH;

  // prologue: async-stage A tile and B tile 0 (8 KB each = 2 x b128 / thread)
  {
    const char* srcA = (const char*)(Pb + (size_t)row0 * CH);
    char* dstA = (char*)&As[0][0];
    cp16_async(dstA + tid * 16, srcA + tid * 16);
    cp16_async(dstA + (tid + 256) * 16, srcA + (tid + 256) * 16);
    const char* srcB = (const char*)Pb;
    char* dstB = (char*)&Bs[0][0][0];
    cp16_async(dstB + tid * 16, srcB + tid * 16);
    cp16_async(dstB + (tid + 256) * 16, srcB + (tid + 256) * 16);
  }
  for (int i = tid; i < RT; i += 256) sqr[i] = sq[b * NPTS + row0 + i];
  for (int i = tid; i < RT * KNN; i += 256) {
    (&topd[0][0])[i] = 3.0e38f;
    (&topi[0][0])[i] = 0;
  }
  wait_async0();
  __syncthreads();

  const int NTILES = NPTS / CT;  // 128
  for (int ct = 0; ct < NTILES; ++ct) {
    int cur = ct & 1;
    int col0 = ct * CT;

    // issue async load of next tile into the other buffer (overlaps compute)
    if (ct + 1 < NTILES) {
      const char* src = (const char*)(Pb + (size_t)(col0 + CT) * CH);
      char* dst = (char*)&Bs[cur ^ 1][0][0];
      cp16_async(dst + tid * 16, src + tid * 16);
      cp16_async(dst + (tid + 256) * 16, src + (tid + 256) * 16);
    }

    // 16 WMMA 16x16 tiles (4x4), 2 per wave, K=64 -> 2 k-steps
#pragma unroll
    for (int i = 0; i < 2; ++i) {
      int t = wave * 2 + i;
      int ms = (t >> 2) * 16, ns = (t & 3) * 16;
      v8f acc = {};
#pragma unroll
      for (int ks = 0; ks < 2; ++ks) {
        v16h a  = frag_ld(&As[0][0], CH, ms, ks * 32, lane);
        v16h bb = frag_ld(&Bs[cur][0][0], CH, ns, ks * 32, lane);
        acc = wmma_f16(a, bb, acc);
      }
      // C layout: VGPR r -> M = r (+8 for lanes>=16), N = lane&15
      int nloc = ns + (lane & 15);
      float sqcv = sq[b * NPTS + col0 + nloc];
      int mbase = ms + ((lane & 16) ? 8 : 0);
#pragma unroll
      for (int r = 0; r < 8; ++r) {
        int mloc = mbase + r;
        Ds[mloc][nloc] = sqr[mloc] + sqcv - 2.0f * acc[r];
      }
    }
    __syncthreads();

    // per-row serialized top-32 insertion (one owner thread per row)
    if (tid < RT) {
      int grow = row0 + tid;
      float* td = topd[tid];
      int*   ti = topi[tid];
      float worst = td[KNN - 1];
      for (int c = 0; c < CT; ++c) {
        int gcol = col0 + c;
        float d = Ds[tid][c];
        if (gcol != grow && d < worst) {
          int p = KNN - 1;
          while (p > 0 && td[p - 1] > d) {
            td[p] = td[p - 1]; ti[p] = ti[p - 1]; --p;
          }
          td[p] = d; ti[p] = gcol;
          worst = td[KNN - 1];
        }
      }
    }
    wait_async0();      // next tile's async copies complete
    __syncthreads();    // Ds consumed + next buffer visible to all waves
  }
  for (int i = tid; i < RT * KNN; i += 256)
    idxOut[((size_t)b * NPTS + row0) * KNN + i] = (&topi[0][0])[i];
}

// ---------------- kernel 3: gather edge feats + GEMM0 (WMMA) + stats --------
__global__ __launch_bounds__(256) void k_gemm0(const _Float16* __restrict__ P16,
                                               const int* __restrict__ idx,
                                               const _Float16* __restrict__ Wh0,
                                               _Float16* __restrict__ y0,
                                               float* __restrict__ sum0,
                                               float* __restrict__ ssq0) {
  __shared__ __align__(16) _Float16 Ws[F1][F0IN];   // 16 KB
  __shared__ __align__(16) _Float16 Es[CT0][F0IN];  // 16 KB
  __shared__ float Ys[F1][CT0];                     // 16 KB
  int tid = threadIdx.x, lane = tid & 31, wave = tid >> 5;
  int g0 = blockIdx.x * CT0;

  stage_async_lds(&Ws[0][0], Wh0, F1 * F0IN * 2, tid);  // overlaps gather below
  {  // build edge-feature columns: [nbr - ctr ; ctr]
    int col = tid >> 2, part = tid & 3;
    int g = g0 + col;
    int p = (g >> 5) & (NPTS - 1);
    int b = g >> 18;
    const _Float16* ctr = P16 + ((size_t)b * NPTS + p) * CH;
    int j = idx[g];
    const _Float16* nbr = P16 + ((size_t)b * NPTS + j) * CH;
    int cb = (part & 1) * 32;
    if (part < 2) {
      for (int c = 0; c < 32; ++c)
        Es[col][cb + c] = (_Float16)((float)nbr[cb + c] - (float)ctr[cb + c]);
    } else {
      for (int c = 0; c < 32; ++c)
        Es[col][64 + cb + c] = ctr[cb + c];
    }
  }
  wait_async0();
  __syncthreads();

  // 16 tiles (M=64 x N=64), 2 per wave, K=128 -> 4 k-steps
#pragma unroll
  for (int i = 0; i < 2; ++i) {
    int t = wave * 2 + i;
    int ms = (t >> 2) * 16, ns = (t & 3) * 16;
    v8f acc = {};
#pragma unroll
    for (int ks = 0; ks < 4; ++ks) {
      v16h a  = frag_ld(&Ws[0][0], F0IN, ms, ks * 32, lane);
      v16h bb = frag_ld(&Es[0][0], F0IN, ns, ks * 32, lane);
      acc = wmma_f16(a, bb, acc);
    }
    int nloc = ns + (lane & 15);
    int mbase = ms + ((lane & 16) ? 8 : 0);
#pragma unroll
    for (int r = 0; r < 8; ++r) Ys[mbase + r][nloc] = acc[r];
  }
  __syncthreads();

  // raw y0 as [col][ch] f16 (contiguous per column for next layer's B load)
  for (int i = tid; i < CT0 * F1; i += 256) {
    int col = i >> 6, c = i & 63;
    y0[(size_t)(g0 + col) * F1 + c] = (_Float16)Ys[c][col];
  }
  if (tid < F1) {  // per-channel partial sums -> global atomics
    float s = 0.f, q = 0.f;
    for (int c2 = 0; c2 < CT0; ++c2) { float v = Ys[tid][c2]; s += v; q += v * v; }
    atomicAdd(&sum0[tid], s);
    atomicAdd(&ssq0[tid], q);
  }
}

// ---------------- BN finalize: scale = g*rstd, shift = b - mean*rstd*g ------
__global__ void k_bnfin(const float* __restrict__ sum, const float* __restrict__ ssq,
                        const float* __restrict__ gw, const float* __restrict__ bw,
                        float* __restrict__ scale, float* __restrict__ shift,
                        int nch, float invCount) {
  int c = blockIdx.x * blockDim.x + threadIdx.x;
  if (c >= nch) return;
  float m = sum[c] * invCount;
  float v = ssq[c] * invCount - m * m;
  float rs = rsqrtf(v + 1e-5f);
  scale[c] = gw[c] * rs;
  shift[c] = bw[c] - m * rs * gw[c];
}

// ---------------- kernel 4: BN0+Leaky fused load -> GEMM1 (WMMA) + stats ----
__global__ __launch_bounds__(256) void k_gemm1(const _Float16* __restrict__ y0,
                                               const _Float16* __restrict__ Wh1,
                                               const float* __restrict__ scale,
                                               const float* __restrict__ shift,
                                               _Float16* __restrict__ y1,
                                               float* __restrict__ sum1,
                                               float* __restrict__ ssq1) {
  __shared__ __align__(16) _Float16 Ws[F2][F1];    // 8 KB
  __shared__ __align__(16) _Float16 Es[CT0][F1];   // 8 KB
  __shared__ float Ys[F2][CT0];                    // 16 KB
  __shared__ float scs[F1], shs[F1];
  int tid = threadIdx.x, lane = tid & 31, wave = tid >> 5;
  int g0 = blockIdx.x * CT0;

  stage_async_lds(&Ws[0][0], Wh1, F2 * F1 * 2, tid);
  if (tid < F1) { scs[tid] = scale[tid]; shs[tid] = shift[tid]; }
  __syncthreads();
  {
    int col = tid >> 2, cb = (tid & 3) * 16;
    const _Float16* src = y0 + (size_t)(g0 + col) * F1;
    for (int c = 0; c < 16; ++c) {
      float v = (float)src[cb + c] * scs[cb + c] + shs[cb + c];
      v = v >= 0.f ? v : NEG_SLOPE * v;
      Es[col][cb + c] = (_Float16)v;
    }
  }
  wait_async0();
  __syncthreads();

#pragma unroll
  for (int i = 0; i < 2; ++i) {
    int t = wave * 2 + i;
    int ms = (t >> 2) * 16, ns = (t & 3) * 16;
    v8f acc = {};
#pragma unroll
    for (int ks = 0; ks < 2; ++ks) {
      v16h a  = frag_ld(&Ws[0][0], F1, ms, ks * 32, lane);
      v16h bb = frag_ld(&Es[0][0], F1, ns, ks * 32, lane);
      acc = wmma_f16(a, bb, acc);
    }
    int nloc = ns + (lane & 15);
    int mbase = ms + ((lane & 16) ? 8 : 0);
#pragma unroll
    for (int r = 0; r < 8; ++r) Ys[mbase + r][nloc] = acc[r];
  }
  __syncthreads();

  for (int i = tid; i < CT0 * F2; i += 256) {
    int col = i >> 6, c = i & 63;
    y1[(size_t)(g0 + col) * F2 + c] = (_Float16)Ys[c][col];
  }
  if (tid < F2) {
    float s = 0.f, q = 0.f;
    for (int c2 = 0; c2 < CT0; ++c2) { float v = Ys[tid][c2]; s += v; q += v * v; }
    atomicAdd(&sum1[tid], s);
    atomicAdd(&ssq1[tid], q);
  }
}

// ---------------- kernel 5: BN1+Leaky fused load -> GEMM2 (WMMA) + stats ----
__global__ __launch_bounds__(256) void k_gemm2(const _Float16* __restrict__ y1,
                                               const _Float16* __restrict__ Wh2,
                                               const float* __restrict__ scale,
                                               const float* __restrict__ shift,
                                               _Float16* __restrict__ y2,
                                               float* __restrict__ sum2,
                                               float* __restrict__ ssq2) {
  __shared__ __align__(16) _Float16 Ws[F3][F2];    // 16 KB
  __shared__ __align__(16) _Float16 Es[CT0][F2];   // 8 KB
  __shared__ float Ys[F3][CT0];                    // 32 KB
  __shared__ float scs[F2], shs[F2];
  int tid = threadIdx.x, lane = tid & 31, wave = tid >> 5;
  int g0 = blockIdx.x * CT0;

  stage_async_lds(&Ws[0][0], Wh2, F3 * F2 * 2, tid);
  if (tid < F2) { scs[tid] = scale[tid]; shs[tid] = shift[tid]; }
  __syncthreads();
  {
    int col = tid >> 2, cb = (tid & 3) * 16;
    const _Float16* src = y1 + (size_t)(g0 + col) * F2;
    for (int c = 0; c < 16; ++c) {
      float v = (float)src[cb + c] * scs[cb + c] + shs[cb + c];
      v = v >= 0.f ? v : NEG_SLOPE * v;
      Es[col][cb + c] = (_Float16)v;
    }
  }
  wait_async0();
  __syncthreads();

  // 32 tiles (M=128 x N=64), 4 per wave, K=64 -> 2 k-steps
#pragma unroll
  for (int i = 0; i < 4; ++i) {
    int t = wave * 4 + i;
    int ms = (t >> 2) * 16, ns = (t & 3) * 16;
    v8f acc = {};
#pragma unroll
    for (int ks = 0; ks < 2; ++ks) {
      v16h a  = frag_ld(&Ws[0][0], F2, ms, ks * 32, lane);
      v16h bb = frag_ld(&Es[0][0], F2, ns, ks * 32, lane);
      acc = wmma_f16(a, bb, acc);
    }
    int nloc = ns + (lane & 15);
    int mbase = ms + ((lane & 16) ? 8 : 0);
#pragma unroll
    for (int r = 0; r < 8; ++r) Ys[mbase + r][nloc] = acc[r];
  }
  __syncthreads();

  for (int i = tid; i < CT0 * F3; i += 256) {
    int col = i >> 7, c = i & 127;
    y2[(size_t)(g0 + col) * F3 + c] = (_Float16)Ys[c][col];
  }
  if (tid < F3) {
    float s = 0.f, q = 0.f;
    for (int c2 = 0; c2 < CT0; ++c2) { float v = Ys[tid][c2]; s += v; q += v * v; }
    atomicAdd(&sum2[tid], s);
    atomicAdd(&ssq2[tid], q);
  }
}

// ---------------- kernel 6: BN2 + Leaky + max over k -> out (B,128,N) ------
__global__ __launch_bounds__(128) void k_final(const _Float16* __restrict__ y2,
                                               const float* __restrict__ scale,
                                               const float* __restrict__ shift,
                                               float* __restrict__ out) {
  int bp = blockIdx.x;            // b*N + p
  int c = threadIdx.x;            // 0..127
  int b = bp >> 13, p = bp & (NPTS - 1);
  float sc = scale[c], sh = shift[c];
  float m = -3.0e38f;
  const _Float16* base = y2 + (size_t)bp * KNN * F3 + c;
#pragma unroll 4
  for (int s = 0; s < KNN; ++s) {
    float v = (float)base[(size_t)s * F3];
    v = v * sc + sh;
    v = v >= 0.f ? v : NEG_SLOPE * v;
    m = fmaxf(m, v);
  }
  out[((size_t)b * F3 + c) * NPTS + p] = m;
}

// ---------------- host launch ----------------
extern "C" void kernel_launch(void* const* d_in, const int* in_sizes, int n_in,
                              void* d_out, int out_size, void* d_ws, size_t ws_size,
                              hipStream_t stream) {
  (void)in_sizes; (void)n_in; (void)out_size; (void)ws_size;
  const float* x  = (const float*)d_in[0];
  const float* W0 = (const float*)d_in[1];
  const float* g0 = (const float*)d_in[2];
  const float* b0 = (const float*)d_in[3];
  const float* W1 = (const float*)d_in[4];
  const float* g1 = (const float*)d_in[5];
  const float* b1 = (const float*)d_in[6];
  const float* W2 = (const float*)d_in[7];
  const float* g2 = (const float*)d_in[8];
  const float* b2 = (const float*)d_in[9];
  float* out = (float*)d_out;

  // workspace carve-up (≈266 MB total)
  char* ws = (char*)d_ws;
  size_t off = 0;
  auto take = [&](size_t bytes) -> char* {
    char* p = ws + off;
    off = (off + bytes + 255) & ~(size_t)255;
    return p;
  };
  _Float16* P16 = (_Float16*)take((size_t)B_ * NPTS * CH * 2);
  float*    sq  = (float*)take((size_t)B_ * NPTS * 4);
  int*      idx = (int*)take((size_t)COLS * 4);
  _Float16* y0  = (_Float16*)take((size_t)COLS * F1 * 2);
  _Float16* y1  = (_Float16*)take((size_t)COLS * F2 * 2);
  _Float16* y2  = (_Float16*)take((size_t)COLS * F3 * 2);
  _Float16* Wh0 = (_Float16*)take((size_t)F1 * F0IN * 2);
  _Float16* Wh1 = (_Float16*)take((size_t)F2 * F1 * 2);
  _Float16* Wh2 = (_Float16*)take((size_t)F3 * F2 * 2);
  float* stats  = (float*)take(4096);
  float* sum0 = stats;       float* ssq0 = stats + 64;
  float* sc0  = stats + 128; float* sh0  = stats + 192;
  float* sum1 = stats + 256; float* ssq1 = stats + 320;
  float* sc1  = stats + 384; float* sh1  = stats + 448;
  float* sum2 = stats + 512; float* ssq2 = stats + 640;
  float* sc2  = stats + 768; float* sh2  = stats + 896;

  const float invCnt = 1.0f / (float)COLS;

  k_zero<<<1, 256, 0, stream>>>(stats);
  k_prep_w<<<(F1 * F0IN + 255) / 256, 256, 0, stream>>>(W0, W1, W2, Wh0, Wh1, Wh2);
  k_prep<<<(B_ * NPTS + 255) / 256, 256, 0, stream>>>(x, P16, sq);
  dim3 ggrid(NPTS / RT, B_);
  k_gram_topk<<<ggrid, 256, 0, stream>>>(P16, sq, idx);
  k_gemm0<<<COLS / CT0, 256, 0, stream>>>(P16, idx, Wh0, y0, sum0, ssq0);
  k_bnfin<<<1, 128, 0, stream>>>(sum0, ssq0, g0, b0, sc0, sh0, F1, invCnt);
  k_gemm1<<<COLS / CT0, 256, 0, stream>>>(y0, Wh1, sc0, sh0, y1, sum1, ssq1);
  k_bnfin<<<1, 128, 0, stream>>>(sum1, ssq1, g1, b1, sc1, sh1, F2, invCnt);
  k_gemm2<<<COLS / CT0, 256, 0, stream>>>(y1, Wh2, sc1, sh1, y2, sum2, ssq2);
  k_bnfin<<<1, 128, 0, stream>>>(sum2, ssq2, g2, b2, sc2, sh2, F3, invCnt);
  k_final<<<B_ * NPTS, 128, 0, stream>>>(y2, sc2, sh2, out);
}
